// MorphSegSeq2SeqModel_60181081752278
// MI455X (gfx1250) — compile-verified, matrix operand order checked by reading, not run on hardware
//
#include <hip/hip_runtime.h>

// ---------------------------------------------------------------------------
// MorphSeg GRU seq2seq decoder for MI455X (gfx1250, wave32, WMMA + TDM)
//   phase 1: x_proj = emb[tokens] @ W_ih^T + b_ih          (WMMA bf16 GEMM)
//   phase 2: 511-step GRU recurrence, W_hh split across 24 WGPs, TDM-staged LDS
//   phase 3: scores = H @ W_out^T + b_out                  (WMMA bf16 GEMM)
// ---------------------------------------------------------------------------

#define VOCAB 52000
#define EMB   768
#define HID   768
#define H3    2304
#define CLS_ID 101

#define GRU_WGS     24
#define ROWS_PER_WG 96          // 24 * 96 = 2304
#define GRU_THREADS 256

typedef __attribute__((ext_vector_type(16))) __bf16 v16bf;
typedef __attribute__((ext_vector_type(8)))  float  v8f;
typedef __attribute__((ext_vector_type(4)))  unsigned int v4u;
typedef __attribute__((ext_vector_type(8)))  int v8i;
typedef __attribute__((ext_vector_type(4)))  int v4i;

// Load a 16-lane-row x 32-K bf16 WMMA fragment from a row-major f32 source.
// Per ISA 7.12.2 (16-bit A 16x32): lane l (half = l>>4) holds two contiguous
// 8-element chunks at kBase + half*8 and kBase + 16 + half*8.
__device__ __forceinline__ v16bf load_frag_bf16(const float* __restrict__ rowPtr,
                                                int kBase, int half) {
  v16bf f;
  const float* p0 = rowPtr + kBase + half * 8;
  const float* p1 = p0 + 16;
#pragma unroll
  for (int e = 0; e < 8; ++e) f[e] = (__bf16)p0[e];
#pragma unroll
  for (int e = 0; e < 8; ++e) f[8 + e] = (__bf16)p1[e];
  return f;
}

// ---------------------------------------------------------------------------
// Phase 1: x_proj[m, n] = dot(emb[tok(m)], W_ih[n]) + b_ih[n]
//   grid = (18, 32), block = 256 (8 waves). One 16x16 tile per wave.
// ---------------------------------------------------------------------------
__global__ void gemm_xproj_kernel(const int* __restrict__ ids,
                                  const float* __restrict__ emb,
                                  const float* __restrict__ W,     // [2304, 768]
                                  const float* __restrict__ bias,  // [2304]
                                  float* __restrict__ C,           // [512, 2304]
                                  int L) {
  const int lane  = threadIdx.x & 31;
  const int wave  = threadIdx.x >> 5;
  const int nTile = blockIdx.x * 8 + wave;       // < 144 by construction
  const int mBase = blockIdx.y * 16;
  const int nBase = nTile * 16;
  const int half  = lane >> 4;
  const int r     = lane & 15;

  const int m   = mBase + r;                      // <= 511
  const int tok = (m == 0) ? CLS_ID : ids[m - 1]; // teacher forcing shift
  const float* aRow = emb + (long)tok * EMB;
  const float* bRow = W + (long)(nBase + r) * EMB;

  v8f acc = {0.f, 0.f, 0.f, 0.f, 0.f, 0.f, 0.f, 0.f};
#pragma unroll 2
  for (int kk = 0; kk < EMB; kk += 32) {
    v16bf a = load_frag_bf16(aRow, kk, half);
    v16bf b = load_frag_bf16(bRow, kk, half);
    acc = __builtin_amdgcn_wmma_f32_16x16x32_bf16(false, a, false, b,
                                                  (short)0, acc, false, false);
  }
  const int   nIdx = nBase + r;
  const float bi   = bias[nIdx];
  if (mBase + 16 <= L) {                          // common case: no guards
#pragma unroll
    for (int v = 0; v < 8; ++v)
      C[(long)(mBase + v + 8 * half) * H3 + nIdx] = acc[v] + bi;
  } else {
#pragma unroll
    for (int v = 0; v < 8; ++v) {
      int mm = mBase + v + 8 * half;
      if (mm < L) C[(long)mm * H3 + nIdx] = acc[v] + bi;
    }
  }
}

// ---------------------------------------------------------------------------
// Phase 2: persistent GRU. 24 WGs, each holds 96 rows of W_hh in LDS (288 KB),
// staged by the Tensor Data Mover. Per step: matvec slice from LDS ->
// publish gh slice -> device barrier -> elementwise GRU update.
// ---------------------------------------------------------------------------
__global__ void gru_kernel(const float* __restrict__ Whh,   // [2304, 768]
                           const float* __restrict__ bhh,   // [2304]
                           const float* __restrict__ enc,   // [768]
                           const float* __restrict__ xproj, // [512, 2304]
                           float* __restrict__ Hout,        // [512, 768]
                           float* __restrict__ gh,          // [2, 2304]
                           unsigned* __restrict__ cnt,      // device barrier
                           int L) {
  extern __shared__ float smem[];
  float* Wsl = smem;                       // ROWS_PER_WG * 768 f32
  float* hb  = smem + ROWS_PER_WG * HID;   // 768 f32 hidden state

  const int tid = threadIdx.x;
  const int wg  = blockIdx.x;

  // --- TDM: DMA the 96x768 f32 W_hh tile into LDS (wave 0 only; TDM ignores
  // EXEC so the wave-uniform branch avoids duplicate DMAs). D# per ISA §8.3/8.4.
  if (tid < 32) {
    const unsigned lds_base = (unsigned)(uintptr_t)Wsl;
    const unsigned long long ga =
        (unsigned long long)(uintptr_t)(Whh + (size_t)wg * ROWS_PER_WG * HID);
    v4u g0;
    g0[0] = 1u;                                           // count=1, user mode
    g0[1] = lds_base;                                     // lds_addr (bytes)
    g0[2] = (unsigned)(ga & 0xFFFFFFFFull);               // global_addr[31:0]
    g0[3] = (unsigned)((ga >> 32) & 0x01FFFFFFull)        // global_addr[56:32]
          | (2u << 30);                                   // type = 2 (image)
    v8i g1;
    g1[0] = (int)(2u << 16);                              // data_size=2 (4B)
    g1[1] = (int)(((unsigned)HID & 0xFFFFu) << 16);       // tensor_dim0 lo16
    g1[2] = (int)(((unsigned)H3 & 0xFFFFu) << 16);        // td0 hi | tensor_dim1 lo
    g1[3] = (int)(((unsigned)HID & 0xFFFFu) << 16);       // td1 hi | tile_dim0=768
    g1[4] = (int)(unsigned)ROWS_PER_WG;                   // tile_dim1=96, tile_dim2=0
    g1[5] = (int)HID;                                     // tensor_dim0_stride lo32
    g1[6] = 0;                                            // strides hi
    g1[7] = 0;
    v4i g2 = {0, 0, 0, 0};                                // 2-D tensor: unused
    v4i g3 = {0, 0, 0, 0};
    v8i g4 = {0, 0, 0, 0, 0, 0, 0, 0};                    // 6-arg form extension
    __builtin_amdgcn_tensor_load_to_lds(g0, g1, g2, g3, g4, 0);
    __builtin_amdgcn_s_wait_tensorcnt(0);
  }
  for (int j = tid; j < HID; j += GRU_THREADS) hb[j] = enc[j];
  __syncthreads();

  const int wave = tid >> 5;
  const int lane = tid & 31;

  for (int t = 0; t < L; ++t) {
    float* ghw = gh + (t & 1) * H3;

    // Cache this lane's 24 h values in registers (reused across 12 rows).
    float hreg[HID / 32];
#pragma unroll
    for (int i = 0; i < HID / 32; ++i) hreg[i] = hb[lane + 32 * i];

    // 8 waves x 12 rows: wave-cooperative dot products from LDS.
#pragma unroll
    for (int rr = 0; rr < 12; ++rr) {
      const int r = wave * 12 + rr;
      const float* wr = Wsl + r * HID;
      float s = 0.f;
#pragma unroll
      for (int i = 0; i < HID / 32; ++i) s = fmaf(wr[lane + 32 * i], hreg[i], s);
#pragma unroll
      for (int off = 16; off; off >>= 1) s += __shfl_xor(s, off, 32);
      if (lane == 0) {
        const int grow = wg * ROWS_PER_WG + r;
        ghw[grow] = s + bhh[grow];
      }
    }
    __threadfence();
    __syncthreads();

    // Device-wide arrive + spin (monotonic counter; gh is double-buffered so
    // one barrier per step is sufficient).
    if (tid == 0) {
      atomicAdd(cnt, 1u);
      const unsigned target = (unsigned)GRU_WGS * (unsigned)(t + 1);
      while (atomicAdd(cnt, 0u) < target) __builtin_amdgcn_s_sleep(4);
    }
    __syncthreads();
    __threadfence();

    // Elementwise GRU cell (redundant in every WG; keeps h local).
    const float* xp = xproj + (long)t * H3;
    float hn[3];
#pragma unroll
    for (int q = 0; q < 3; ++q) {
      const int j = tid + q * GRU_THREADS;   // 3 * 256 = 768
      const float rg = 1.f / (1.f + __expf(-(xp[j] + ghw[j])));
      const float zg = 1.f / (1.f + __expf(-(xp[HID + j] + ghw[HID + j])));
      const float ng = tanhf(xp[2 * HID + j] + rg * ghw[2 * HID + j]);
      hn[q] = (1.f - zg) * ng + zg * hb[j];
    }
#pragma unroll
    for (int q = 0; q < 3; ++q) {
      const int j = tid + q * GRU_THREADS;
      hb[j] = hn[q];
      if (wg == 0) Hout[(long)t * HID + j] = hn[q];
    }
    __syncthreads();
  }
}

// ---------------------------------------------------------------------------
// Phase 3: scores[m, n] = dot(H[m], W_out[n]) + b_out[n]
//   grid = (407, 32), block = 256 (8 waves). One 16x16 tile per wave.
//   Memory-bound (160 MB W_out @ 23.3 TB/s) -> bf16 WMMA is past sufficient.
// ---------------------------------------------------------------------------
__global__ void gemm_out_kernel(const float* __restrict__ A,     // [512, 768]
                                const float* __restrict__ W,     // [52000, 768]
                                const float* __restrict__ bias,  // [52000]
                                float* __restrict__ C,           // [L, 52000]
                                int L) {
  const int lane  = threadIdx.x & 31;
  const int wave  = threadIdx.x >> 5;
  const int nTile = blockIdx.x * 8 + wave;
  if (nTile >= VOCAB / 16) return;               // 3250 tiles
  const int mBase = blockIdx.y * 16;
  const int nBase = nTile * 16;
  const int half  = lane >> 4;
  const int r     = lane & 15;

  const float* aRow = A + (long)(mBase + r) * HID;
  const float* bRow = W + (long)(nBase + r) * HID;

  // Warm the streaming path (lowers to global_prefetch_b8).
  __builtin_prefetch(bRow + 256, 0, 1);
  __builtin_prefetch(bRow + 512, 0, 1);

  v8f acc = {0.f, 0.f, 0.f, 0.f, 0.f, 0.f, 0.f, 0.f};
#pragma unroll 2
  for (int kk = 0; kk < HID; kk += 32) {
    v16bf a = load_frag_bf16(aRow, kk, half);
    v16bf b = load_frag_bf16(bRow, kk, half);
    acc = __builtin_amdgcn_wmma_f32_16x16x32_bf16(false, a, false, b,
                                                  (short)0, acc, false, false);
  }
  const int   nIdx = nBase + r;
  const float bi   = bias[nIdx];
  if (mBase + 16 <= L) {                          // common case: no guards
#pragma unroll
    for (int v = 0; v < 8; ++v)
      C[(long)(mBase + v + 8 * half) * VOCAB + nIdx] = acc[v] + bi;
  } else {
#pragma unroll
    for (int v = 0; v < 8; ++v) {
      int m = mBase + v + 8 * half;
      if (m < L) C[(long)m * VOCAB + nIdx] = acc[v] + bi;
    }
  }
}

__global__ void init_cnt_kernel(unsigned* c) {
  if (threadIdx.x == 0) *c = 0u;
}

// ---------------------------------------------------------------------------
extern "C" void kernel_launch(void* const* d_in, const int* in_sizes, int n_in,
                              void* d_out, int out_size, void* d_ws, size_t ws_size,
                              hipStream_t stream) {
  const int*   ids  = (const int*)d_in[0];    // xform_ids [1,512]
  //            d_in[1] = xform_mask (L derived from out_size instead)
  const float* enc  = (const float*)d_in[2];  // enc_hs [1,768]
  const float* emb  = (const float*)d_in[3];  // emb_table [52000,768]
  const float* Wih  = (const float*)d_in[4];  // [2304,768]
  const float* Whh  = (const float*)d_in[5];  // [2304,768]
  const float* bih  = (const float*)d_in[6];  // [2304]
  const float* bhh  = (const float*)d_in[7];  // [2304]
  const float* Wout = (const float*)d_in[8];  // [52000,768]
  const float* bout = (const float*)d_in[9];  // [52000]
  float* out = (float*)d_out;

  int L = out_size / VOCAB;                   // 511 for the reference inputs
  if (L > 512) L = 512;
  if (L < 1)  L = 1;

  // Workspace layout (floats): xproj | H | gh double buffer | counter
  float*    xproj = (float*)d_ws;
  float*    Hbuf  = xproj + (size_t)512 * H3;
  float*    ghbuf = Hbuf + (size_t)512 * HID;
  unsigned* cnt   = (unsigned*)(ghbuf + 2 * H3);

  init_cnt_kernel<<<1, 32, 0, stream>>>(cnt);

  // Phase 1: teacher-forced embedding gather + input projection GEMM.
  gemm_xproj_kernel<<<dim3(18, 32), dim3(256), 0, stream>>>(
      ids, emb, Wih, bih, xproj, L);

  // Phase 2: persistent GRU across 24 WGPs, W_hh slice TDM-staged into LDS.
  const size_t gruShmem = (size_t)(ROWS_PER_WG * HID + HID) * sizeof(float);
  gru_kernel<<<GRU_WGS, GRU_THREADS, gruShmem, stream>>>(
      Whh, bhh, enc, xproj, Hbuf, ghbuf, cnt, L);

  // Phase 3: vocabulary projection GEMM.
  gemm_out_kernel<<<dim3(407, 32), dim3(256), 0, stream>>>(
      Hbuf, Wout, bout, out, L);
}
